// MoELayer_76115410420405
// MI455X (gfx1250) — matrix operand order repaired and checked by reading, not compiled, not run on hardware
//
#include <hip/hip_runtime.h>
#include <math.h>

// ---------------- problem constants ----------------
#define BB 2
#define SS 2048
#define DD 1024
#define EE 8
#define KK 2
#define HH 4096
#define NT (BB * SS)        // 4096 tokens
#define NP (NT * KK)        // 8192 (token,slot) pairs

typedef __attribute__((ext_vector_type(16))) __bf16 v16bf;
typedef __attribute__((ext_vector_type(8)))  __bf16 v8bf;
typedef __attribute__((ext_vector_type(8)))  float  v8f;

// ---------------- workspace layout (bytes) ----------------
#define WS_COUNTS  0
#define WS_LISTS   256
#define WS_WGT     (WS_LISTS + EE * NT * 4)              // 131328
#define WS_XBF     (WS_WGT + NP * 4)                     // 164096
#define WS_HBUF    (WS_XBF + (size_t)NT * DD * 2)        // 8552704
#define WS_PARTIAL (WS_HBUF + (size_t)NP * HH * 2)       // 75661568
// total 109216000 bytes (~104.2 MB)

// LDS column strides (pad 8 elems -> 16B so 16 column-lanes hit distinct banks)
#define LDK1 (DD + 8)   // 1032
#define LDK2 (HH + 8)   // 4104

// ============ helpers ============

// Build a 16-element bf16 fragment from two 16B chunks (8+8 elems).
// A-matrix (16-bit 16x32 MxK) ISA layout: lane m=lane&15;
//   lanes 0-15 : elems 0..7 = K0..7,  elems 8..15 = K16..23
//   lanes16-31 : elems 0..7 = K8..15, elems 8..15 = K24..31
// B-matrix (32x16 KxN): lane n=lane&15 is column; same K order as A:
//   lanes 0-15 : K0..7, K16..23; lanes 16-31 : K8..15, K24..31.
__device__ __forceinline__ void ld_frag(v16bf& a, const __bf16* p_lo, const __bf16* p_hi) {
  const v8bf lo = *(const v8bf*)(p_lo);
  const v8bf hi = *(const v8bf*)(p_hi);
#pragma unroll
  for (int i = 0; i < 8; ++i) { a[i] = lo[i]; a[8 + i] = hi[i]; }
}

__device__ __forceinline__ v8bf cvt8(const float4 q0, const float4 q1) {
  v8bf v;
  v[0]=(__bf16)q0.x; v[1]=(__bf16)q0.y; v[2]=(__bf16)q0.z; v[3]=(__bf16)q0.w;
  v[4]=(__bf16)q1.x; v[5]=(__bf16)q1.y; v[6]=(__bf16)q1.z; v[7]=(__bf16)q1.w;
  return v;
}

#define WMMA_BF16(A, B, C) \
  __builtin_amdgcn_wmma_f32_16x16x32_bf16(false, (A), false, (B), (short)0, (C), false, false)

// ============ kernel 0: init counters ============
__global__ void moe_init_kernel(int* __restrict__ counts) {
  if (threadIdx.x < EE) counts[threadIdx.x] = 0;
}

// ============ kernel 1: gate + routing (one wave per token) ============
// Also emits the bf16 copy of x (row is already streaming through registers).
__global__ __launch_bounds__(256) void moe_gate_kernel(
    const float* __restrict__ x,       // [NT, D]
    const float* __restrict__ gate_w,  // [E, D]
    const float* __restrict__ gate_b,  // [E]
    int* __restrict__ counts,          // [E]
    int* __restrict__ lists,           // [E, NT]
    float* __restrict__ pair_wgt,      // [NP]
    float* __restrict__ out_sel,       // [NT*K] tail of d_out
    __bf16* __restrict__ xbf)          // [NT, D] bf16 copy of x
{
  __shared__ float s_gw[EE * DD];  // 32KB
  for (int i = threadIdx.x; i < EE * DD; i += 256) s_gw[i] = gate_w[i];
  __syncthreads();

  const int wave = threadIdx.x >> 5;
  const int lane = threadIdx.x & 31;
  const int t = blockIdx.x * 8 + wave;
  const float* xr = x + (size_t)t * DD;
  __bf16* xbr = xbf + (size_t)t * DD;

  float acc[EE];
#pragma unroll
  for (int e = 0; e < EE; ++e) acc[e] = 0.0f;

  for (int i = 0; i < DD; i += 32) {
    const float xv = xr[i + lane];
    xbr[i + lane] = (__bf16)xv;     // free bf16 cast of x on the way through
#pragma unroll
    for (int e = 0; e < EE; ++e) acc[e] += xv * s_gw[e * DD + i + lane];
  }
#pragma unroll
  for (int e = 0; e < EE; ++e) {
    float v = acc[e];
#pragma unroll
    for (int off = 16; off >= 1; off >>= 1) v += __shfl_xor(v, off, 32);
    acc[e] = v;
  }

  if (lane == 0) {
    float lg[EE];
#pragma unroll
    for (int e = 0; e < EE; ++e) lg[e] = acc[e] + gate_b[e];
    int e0 = 0;
#pragma unroll
    for (int e = 1; e < EE; ++e) if (lg[e] > lg[e0]) e0 = e;   // ties -> lowest idx
    int e1 = (e0 == 0) ? 1 : 0;
#pragma unroll
    for (int e = 0; e < EE; ++e) if (e != e0 && lg[e] > lg[e1]) e1 = e;
    const float r = __expf(lg[e1] - lg[e0]);   // <= 1
    const float w0 = 1.0f / (1.0f + r);
    const float w1 = r * w0;

    const int p0 = t * 2, p1 = t * 2 + 1;
    pair_wgt[p0] = w0;
    pair_wgt[p1] = w1;
    const int pos0 = atomicAdd(&counts[e0], 1);
    lists[e0 * NT + pos0] = p0;
    const int pos1 = atomicAdd(&counts[e1], 1);
    lists[e1 * NT + pos1] = p1;
    out_sel[p0] = (float)e0;
    out_sel[p1] = (float)e1;
  }
}

// ============ kernel 2: h = gelu(x @ w1^T + b1), bf16 out ============
// grid: x = H/32 col-groups, z = E. block = 256 (8 waves).
// Block stages w1 slice [32 cols x 1024 K] into LDS (bf16, once), then each
// wave iterates the expert's 32-row pair tiles (stride 8), K-loop from LDS.
__global__ __launch_bounds__(256) void moe_ffn1_kernel(
    const __bf16* __restrict__ xbf,  // [NT, D] bf16
    const float* __restrict__ w1,    // [E, H, D]
    const float* __restrict__ b1,    // [E, H]
    const int* __restrict__ counts,
    const int* __restrict__ lists,   // [E, NT]
    __bf16* __restrict__ hbuf)       // [NP, H]
{
  extern __shared__ __bf16 smem[];   // [32][LDK1]
  const int e = blockIdx.z;
  const int cnt = counts[e];
  if (cnt == 0) return;
  const int colbase = blockIdx.x * 32;

  // ---- stage weights: 32 cols x 1024 K, fp32 -> bf16 (read once from HBM) ----
  const float* w1e = w1 + (size_t)e * HH * DD + (size_t)colbase * DD;
  for (int ch = threadIdx.x; ch < 32 * (DD / 8); ch += 256) {
    const int c = ch >> 7;            // / (DD/8)
    const int kc = (ch & 127) * 8;
    const float4 q0 = *(const float4*)(w1e + (size_t)c * DD + kc);
    const float4 q1 = *(const float4*)(w1e + (size_t)c * DD + kc + 4);
    *(v8bf*)(smem + c * LDK1 + kc) = cvt8(q0, q1);
  }
  __syncthreads();

  const int wave = threadIdx.x >> 5;
  const int lane = threadIdx.x & 31;
  const int m = lane & 15;
  const int half = lane >> 4;
  const int koffA = half * 8;
  const int koffB = half * 8;

  const __bf16* bl0 = smem + m * LDK1 + koffB;          // B tile0 column
  const __bf16* bl1 = smem + (16 + m) * LDK1 + koffB;   // B tile1 column
  const float bias0 = b1[(size_t)e * HH + colbase + m];
  const float bias1 = b1[(size_t)e * HH + colbase + 16 + m];

  const int ntiles = (cnt + 31) >> 5;
  for (int pt = wave; pt < ntiles; pt += 8) {
    const int tileM = pt * 32;
    const int i0 = tileM + m, i1 = tileM + 16 + m;
    const int p0 = (i0 < cnt) ? lists[e * NT + i0] : 0;
    const int p1 = (i1 < cnt) ? lists[e * NT + i1] : 0;
    const __bf16* ar0 = xbf + (size_t)(p0 >> 1) * DD;
    const __bf16* ar1 = xbf + (size_t)(p1 >> 1) * DD;

    v8f c00 = {}, c01 = {}, c10 = {}, c11 = {};
#pragma unroll 2
    for (int kb = 0; kb < DD; kb += 32) {
      v16bf a0, a1, b0, b1v;
      ld_frag(a0, ar0 + kb + koffA, ar0 + kb + 16 + koffA);
      ld_frag(a1, ar1 + kb + koffA, ar1 + kb + 16 + koffA);
      ld_frag(b0, bl0 + kb, bl0 + kb + 16);
      ld_frag(b1v, bl1 + kb, bl1 + kb + 16);
      __builtin_prefetch(ar0 + kb + 512, 0, 3);   // WGP-scope prefetch of A stream
      __builtin_prefetch(ar1 + kb + 512, 0, 3);
      c00 = WMMA_BF16(a0, b0, c00);
      c01 = WMMA_BF16(a0, b1v, c01);
      c10 = WMMA_BF16(a1, b0, c10);
      c11 = WMMA_BF16(a1, b1v, c11);
    }

    // epilogue: +b1, exact GELU, store bf16.
    // C layout: elem r -> row r + half*8 (of 16-row subtile), col = lane&15.
#pragma unroll
    for (int s = 0; s < 2; ++s) {
      const v8f cs0 = s ? c10 : c00;
      const v8f cs1 = s ? c11 : c01;
#pragma unroll
      for (int r = 0; r < 8; ++r) {
        const int row = r + half * 8;
        const int pair = __shfl(s ? p1 : p0, row, 32);
        if (tileM + s * 16 + row < cnt) {
          const size_t pr = (size_t)pair * HH;
          float v0 = cs0[r] + bias0;
          float v1 = cs1[r] + bias1;
          v0 = 0.5f * v0 * (1.0f + erff(v0 * 0.70710678118654752f));
          v1 = 0.5f * v1 * (1.0f + erff(v1 * 0.70710678118654752f));
          hbuf[pr + colbase + m] = (__bf16)v0;
          hbuf[pr + colbase + 16 + m] = (__bf16)v1;
        }
      }
    }
  }
}

// ============ kernel 3: partial = wgt * (h @ w2^T + b2) ============
// grid: x = D/32 col-groups, z = E. block = 256 (8 waves).
// LDS: w2 slice [32 cols x 4096 K] bf16 = 257KB (dynamic, fits 320KB WGP LDS).
__global__ __launch_bounds__(256) void moe_ffn2_kernel(
    const __bf16* __restrict__ hbuf,  // [NP, H] bf16
    const float* __restrict__ w2,     // [E, D, H]
    const float* __restrict__ b2,     // [E, D]
    const int* __restrict__ counts,
    const int* __restrict__ lists,
    const float* __restrict__ pair_wgt,
    float* __restrict__ partial)      // [NP, D]
{
  extern __shared__ __bf16 smem[];   // [32][LDK2]
  const int e = blockIdx.z;
  const int cnt = counts[e];
  if (cnt == 0) return;
  const int colbase = blockIdx.x * 32;

  // ---- stage weights: 32 cols x 4096 K, fp32 -> bf16 (read once from HBM) ----
  const float* w2e = w2 + (size_t)e * DD * HH + (size_t)colbase * HH;
  for (int ch = threadIdx.x; ch < 32 * (HH / 8); ch += 256) {
    const int c = ch >> 9;            // / (HH/8)
    const int kc = (ch & 511) * 8;
    const float4 q0 = *(const float4*)(w2e + (size_t)c * HH + kc);
    const float4 q1 = *(const float4*)(w2e + (size_t)c * HH + kc + 4);
    *(v8bf*)(smem + c * LDK2 + kc) = cvt8(q0, q1);
  }
  __syncthreads();

  const int wave = threadIdx.x >> 5;
  const int lane = threadIdx.x & 31;
  const int m = lane & 15;
  const int half = lane >> 4;
  const int koffA = half * 8;
  const int koffB = half * 8;

  const __bf16* bl0 = smem + m * LDK2 + koffB;
  const __bf16* bl1 = smem + (16 + m) * LDK2 + koffB;
  const float bias0 = b2[(size_t)e * DD + colbase + m];
  const float bias1 = b2[(size_t)e * DD + colbase + 16 + m];

  const int ntiles = (cnt + 31) >> 5;
  for (int pt = wave; pt < ntiles; pt += 8) {
    const int tileM = pt * 32;
    const int i0 = tileM + m, i1 = tileM + 16 + m;
    const int p0 = (i0 < cnt) ? lists[e * NT + i0] : 0;
    const int p1 = (i1 < cnt) ? lists[e * NT + i1] : 0;
    const float wg0 = (i0 < cnt) ? pair_wgt[p0] : 0.0f;
    const float wg1 = (i1 < cnt) ? pair_wgt[p1] : 0.0f;
    const __bf16* ar0 = hbuf + (size_t)p0 * HH;
    const __bf16* ar1 = hbuf + (size_t)p1 * HH;

    v8f c00 = {}, c01 = {}, c10 = {}, c11 = {};
#pragma unroll 2
    for (int kb = 0; kb < HH; kb += 32) {
      v16bf a0, a1, b0, b1v;
      ld_frag(a0, ar0 + kb + koffA, ar0 + kb + 16 + koffA);
      ld_frag(a1, ar1 + kb + koffA, ar1 + kb + 16 + koffA);
      ld_frag(b0, bl0 + kb, bl0 + kb + 16);
      ld_frag(b1v, bl1 + kb, bl1 + kb + 16);
      __builtin_prefetch(ar0 + kb + 512, 0, 3);
      __builtin_prefetch(ar1 + kb + 512, 0, 3);
      c00 = WMMA_BF16(a0, b0, c00);
      c01 = WMMA_BF16(a0, b1v, c01);
      c10 = WMMA_BF16(a1, b0, c10);
      c11 = WMMA_BF16(a1, b1v, c11);
    }

#pragma unroll
    for (int s = 0; s < 2; ++s) {
      const v8f cs0 = s ? c10 : c00;
      const v8f cs1 = s ? c11 : c01;
#pragma unroll
      for (int r = 0; r < 8; ++r) {
        const int row = r + half * 8;
        const int pair = __shfl(s ? p1 : p0, row, 32);
        const float wk = __shfl(s ? wg1 : wg0, row, 32);
        if (tileM + s * 16 + row < cnt) {
          const size_t pr = (size_t)pair * DD;
          partial[pr + colbase + m] = (cs0[r] + bias0) * wk;
          partial[pr + colbase + 16 + m] = (cs1[r] + bias1) * wk;
        }
      }
    }
  }
}

// ============ kernel 4: combine ============
__global__ __launch_bounds__(256) void moe_combine_kernel(
    const float* __restrict__ partial, float* __restrict__ out) {
  const size_t i = (size_t)blockIdx.x * 256 + threadIdx.x;  // < NT*D
  const size_t t = i / DD;
  const size_t d = i - t * DD;
  out[i] = partial[(t * 2) * DD + d] + partial[(t * 2 + 1) * DD + d];
}

// ============ launch ============
extern "C" void kernel_launch(void* const* d_in, const int* in_sizes, int n_in,
                              void* d_out, int out_size, void* d_ws, size_t ws_size,
                              hipStream_t stream) {
  const float* x      = (const float*)d_in[0];
  const float* gate_w = (const float*)d_in[1];
  const float* gate_b = (const float*)d_in[2];
  const float* w1     = (const float*)d_in[3];
  const float* b1     = (const float*)d_in[4];
  const float* w2     = (const float*)d_in[5];
  const float* b2     = (const float*)d_in[6];

  char* ws = (char*)d_ws;
  int*    counts   = (int*)(ws + WS_COUNTS);
  int*    lists    = (int*)(ws + WS_LISTS);
  float*  pair_wgt = (float*)(ws + WS_WGT);
  __bf16* xbf      = (__bf16*)(ws + WS_XBF);
  __bf16* hbuf     = (__bf16*)(ws + WS_HBUF);
  float*  partial  = (float*)(ws + WS_PARTIAL);

  float* out  = (float*)d_out;            // [NT, D]
  float* osel = out + (size_t)NT * DD;    // [NT, K] selected experts (as float)

  moe_init_kernel<<<1, 32, 0, stream>>>(counts);
  moe_gate_kernel<<<NT / 8, 256, 0, stream>>>(x, gate_w, gate_b, counts, lists,
                                              pair_wgt, osel, xbf);
  moe_ffn1_kernel<<<dim3(HH / 32, 1, EE), 256, 32 * LDK1 * 2, stream>>>(
      xbf, w1, b1, counts, lists, hbuf);
  moe_ffn2_kernel<<<dim3(DD / 32, 1, EE), 256, 32 * LDK2 * 2, stream>>>(
      hbuf, w2, b2, counts, lists, pair_wgt, partial);
  moe_combine_kernel<<<(NT * DD) / 256, 256, 0, stream>>>(partial, out);
}
